// self_distillation_54348516163877
// MI455X (gfx1250) — matrix operand order repaired and checked by reading, not compiled, not run on hardware
//
#include <hip/hip_runtime.h>
#include <hip/hip_bf16.h>

// ---------------------------------------------------------------------------
// NT-Xent (SimCLR) loss for MI455X / gfx1250.
//
// Roofline: GEMM1 streams 252 MB of fp32 W1 (read exactly once) -> memory
// bound (~10.8 us @ 23.3 TB/s). Matrix math: v_wmma_f32_16x16x32_bf16 with a
// hi+lo bf16 split (3 WMMAs / K-step) for ~fp32 accuracy.
//
// Structure:
//  * Activations (X, H) are split into bf16 hi/lo planes ONCE when produced;
//    A-fragments are then pure b128 loads (no per-K-step conversion VALU).
//  * Each wave owns 16 distinct W columns x all 256 rows, so every weight
//    element is converted fp32->bf16(hi,lo) exactly once per HBM read.
//  * Row tiles processed in groups of 4: 16 b128 loads batched per group,
//    WMMAs issued in 3 interleaved passes so dependent WMMAs on the same
//    accumulator are separated by 3 independent WMMAs (fills the D->C hazard
//    co-exec slots, removes v_nop padding; deepens the load clause).
//  * Deterministic split-K x4 (disjoint partial buffers + fixed-order
//    reduction; no float atomics) to spread W streaming over 128 blocks.
//  * global_prefetch on the weight stream.
// ---------------------------------------------------------------------------

typedef __attribute__((ext_vector_type(16))) __bf16 v16bf;
typedef __attribute__((ext_vector_type(8)))  float  v8f;

union BF16x16 { v16bf v; __bf16 e[16]; };
union BFLD    { v16bf v; uint4  q[2];  };

static constexpr int MROWS = 256;     // 2 * batch
static constexpr int DIN   = 15360;   // 512*5*6
static constexpr int N1    = 4096;
static constexpr int N2    = 2048;
static constexpr int SPLITK = 4;
static constexpr float T_INV = 2.0f;  // 1 / temperature(0.5)

__device__ __forceinline__ void splitStore(float x, BF16x16& H, BF16x16& L, int i) {
  __bf16 h = (__bf16)x;
  H.e[i] = h;
  L.e[i] = (__bf16)(x - (float)h);   // residual: ~8 extra mantissa bits
}

// B-matrix 32x16 bf16 fragment: lane = 16*half + n, column n = one row of Wt,
// e0..e7 <- K = kbase + half*8 + {0..7}, e8..e15 <- K = kbase + 16 + half*8 + {0..7}.
__device__ __forceinline__ void loadB_split(const float* __restrict__ p, float scale,
                                            v16bf& hi, v16bf& lo) {
  BF16x16 H, L;
#pragma unroll
  for (int i = 0; i < 8; ++i) {
    splitStore(p[i]      * scale, H, L, i);
    splitStore(p[i + 16] * scale, H, L, i + 8);
  }
  hi = H.v; lo = L.v;
}

// A-matrix 16x32 bf16 fragment from a pre-split bf16 plane:
// lane = 16*half + m; e0..e7 <- p[0..7], e8..e15 <- p[16..23]  (p is bf16*)
__device__ __forceinline__ v16bf loadA_plane(const __bf16* __restrict__ p) {
  BFLD u;
  u.q[0] = *(const uint4*)(p);        // 8 bf16 = 16B
  u.q[1] = *(const uint4*)(p + 16);   // 8 bf16 at +32B
  return u.v;
}

// fp32 A fragment with scale + on-the-fly split (used only by tiny Gram GEMM)
__device__ __forceinline__ void loadA_split_f32(const float* __restrict__ p, float scale,
                                                v16bf& hi, v16bf& lo) {
  BF16x16 H, L;
#pragma unroll
  for (int i = 0; i < 8; ++i) {
    splitStore(p[i]      * scale, H, L, i);
    splitStore(p[i + 16] * scale, H, L, i + 8);
  }
  hi = H.v; lo = L.v;
}

__device__ __forceinline__ v8f wmma1(v8f acc, v16bf a, v16bf b) {
  return __builtin_amdgcn_wmma_f32_16x16x32_bf16(false, a, false, b, (short)0, acc,
                                                 false, false);
}

// acc += Ahi*Bhi + Ahi*Blo + Alo*Bhi  (lo*lo negligible) — used by Gram GEMM
__device__ __forceinline__ v8f wmma3(v8f acc, v16bf ah, v16bf al, v16bf bh, v16bf bl) {
  acc = wmma1(acc, al, bh);
  acc = wmma1(acc, ah, bl);
  acc = wmma1(acc, ah, bh);
  return acc;
}

// ---------------------------------------------------------------------------
// Kernel 1: build X hi/lo bf16 planes [256, 15360];
// rows 0..127 = ct_out, rows 128..255 = mct_out with w-dim (6) reversed.
// ---------------------------------------------------------------------------
__global__ __launch_bounds__(256) void build_X_planes(const float* __restrict__ ct,
                                                      const float* __restrict__ mct,
                                                      __bf16* __restrict__ Xhi,
                                                      __bf16* __restrict__ Xlo) {
  const int idx = blockIdx.x * 256 + threadIdx.x;   // covers 256*15360
  const int row = idx / DIN;
  const int e   = idx - row * DIN;
  float v;
  if (row < 128) {
    v = ct[row * DIN + e];
  } else {
    const int w = e % 6;
    v = mct[(row - 128) * DIN + (e - w) + (5 - w)];
  }
  __bf16 h = (__bf16)v;
  Xhi[idx] = h;
  Xlo[idx] = (__bf16)(v - (float)h);
}

// ---------------------------------------------------------------------------
// Kernel 2/4: split-K GEMM.  part[ksplit][256][N] = A(planes)[256,K] @ Wt[N,K]^T
// block = 256 thr (8 waves); wave owns 16 cols x ALL 256 rows (16 row tiles,
// 128 acc VGPRs).  grid = (N/128, SPLITK)
// ---------------------------------------------------------------------------
__global__ __launch_bounds__(256) void wmma_gemm_splitk(
    const __bf16* __restrict__ Ahi, const __bf16* __restrict__ Alo,
    const float* __restrict__ Wt, float* __restrict__ part,
    int K, int N, int kChunk) {
  const int lane = threadIdx.x & 31;
  const int wave = threadIdx.x >> 5;
  const int ml   = lane & 15;
  const int hlf  = lane >> 4;

  const int colBase = blockIdx.x * 128 + wave * 16;   // this wave's 16 columns
  const int kBeg = blockIdx.y * kChunk;
  const int kEnd = kBeg + kChunk;
  float* __restrict__ out = part + (size_t)blockIdx.y * MROWS * N;

  const float*  bP   = Wt  + (size_t)(colBase + ml) * K + hlf * 8;
  const __bf16* aHi0 = Ahi + (size_t)ml * K + hlf * 8;
  const __bf16* aLo0 = Alo + (size_t)ml * K + hlf * 8;

  v8f zero;
#pragma unroll
  for (int t = 0; t < 8; ++t) zero[t] = 0.0f;
  v8f acc[16];
#pragma unroll
  for (int rt = 0; rt < 16; ++rt) acc[rt] = zero;

  for (int k = kBeg; k < kEnd; k += 32) {
    __builtin_prefetch(bP + k + 256, 0, 1);   // weight stream, 8 K-steps ahead
    v16bf bh, bl;
    loadB_split(bP + k, 1.0f, bh, bl);        // only conversion in hot loop

    // 4 groups x 4 row tiles: batch 16 b128 loads, then 12 WMMAs in three
    // interleaved passes (3 independent WMMAs between dependent ones).
#pragma unroll
    for (int g = 0; g < 4; ++g) {
      v16bf ah[4], al[4];
#pragma unroll
      for (int t = 0; t < 4; ++t) {
        const size_t ao = (size_t)((g * 4 + t) * 16) * K + k;
        ah[t] = loadA_plane(aHi0 + ao);
        al[t] = loadA_plane(aLo0 + ao);
      }
#pragma unroll
      for (int t = 0; t < 4; ++t) acc[g * 4 + t] = wmma1(acc[g * 4 + t], al[t], bh);
#pragma unroll
      for (int t = 0; t < 4; ++t) acc[g * 4 + t] = wmma1(acc[g * 4 + t], ah[t], bl);
#pragma unroll
      for (int t = 0; t < 4; ++t) acc[g * 4 + t] = wmma1(acc[g * 4 + t], ah[t], bh);
    }
  }

  // C/D layout: lane(0..15): N=ml, M=v; lane(16..31): N=ml, M=v+8
  const int col = colBase + ml;
#pragma unroll
  for (int rt = 0; rt < 16; ++rt) {
    const int r0 = rt * 16 + hlf * 8;
#pragma unroll
    for (int v = 0; v < 8; ++v)
      out[(size_t)(r0 + v) * N + col] = acc[rt][v];
  }
}

// ---------------------------------------------------------------------------
// Kernel 3/5: deterministic split-K reduction + bias + ReLU; emits bf16 hi/lo
// planes (for next GEMM's A operand) and/or fp32 (for norms/Gram).
// ---------------------------------------------------------------------------
__global__ __launch_bounds__(256) void reduce_bias_relu(
    const float* __restrict__ part, const float* __restrict__ bias,
    __bf16* __restrict__ Ohi, __bf16* __restrict__ Olo,
    float* __restrict__ Ofp, int N) {
  const int idx = blockIdx.x * 256 + threadIdx.x;   // covers 256*N
  const int col = idx % N;
  float s = bias[col];
#pragma unroll
  for (int sp = 0; sp < SPLITK; ++sp)               // fixed order: deterministic
    s += part[(size_t)sp * MROWS * N + idx];
  s = fmaxf(s, 0.0f);
  if (Ofp) Ofp[idx] = s;
  if (Ohi) {
    __bf16 h = (__bf16)s;
    Ohi[idx] = h;
    Olo[idx] = (__bf16)(s - (float)h);
  }
}

// ---------------------------------------------------------------------------
// Kernel 6: inverse row norms (clamped) of R[256, 2048]
// ---------------------------------------------------------------------------
__global__ __launch_bounds__(256) void row_inv_norms(const float* __restrict__ R,
                                                     float* __restrict__ inv) {
  __shared__ float red[256];
  const int r = blockIdx.x;
  float s = 0.0f;
  for (int c = threadIdx.x; c < N2; c += 256) {
    float v = R[(size_t)r * N2 + c];
    s += v * v;
  }
  red[threadIdx.x] = s;
  __syncthreads();
  for (int st = 128; st > 0; st >>= 1) {
    if (threadIdx.x < st) red[threadIdx.x] += red[threadIdx.x + st];
    __syncthreads();
  }
  if (threadIdx.x == 0) inv[r] = 1.0f / fmaxf(sqrtf(red[0]), 1e-8f);
}

// ---------------------------------------------------------------------------
// Kernel 7: sim = RN @ RN^T (256x256, K=2048) — tiny; normalization fused
// into fragment loads.  grid = (4, 2); wave -> 16 rows x 64 cols.
// ---------------------------------------------------------------------------
__global__ __launch_bounds__(256) void wmma_gram(const float* __restrict__ R,
                                                 const float* __restrict__ inv,
                                                 float* __restrict__ sim) {
  const int lane = threadIdx.x & 31;
  const int wave = threadIdx.x >> 5;
  const int ml   = lane & 15;
  const int hlf  = lane >> 4;

  const int rowTile = blockIdx.y * 128 + wave * 16;
  const int colBase = blockIdx.x * 64;

  const float sA = inv[rowTile + ml];
  const float* aP = R + (size_t)(rowTile + ml) * N2 + hlf * 8;
  const float* bP[4];
  float sB[4];
#pragma unroll
  for (int j = 0; j < 4; ++j) {
    const int br = colBase + j * 16 + ml;
    sB[j] = inv[br];
    bP[j] = R + (size_t)br * N2 + hlf * 8;
  }

  v8f zero;
#pragma unroll
  for (int t = 0; t < 8; ++t) zero[t] = 0.0f;
  v8f acc[4] = {zero, zero, zero, zero};

  for (int k = 0; k < N2; k += 32) {
    v16bf ah, al;
    loadA_split_f32(aP + k, sA, ah, al);
#pragma unroll
    for (int j = 0; j < 4; ++j) {
      v16bf bh, bl;
      loadB_split(bP[j] + k, sB[j], bh, bl);
      acc[j] = wmma3(acc[j], ah, al, bh, bl);
    }
  }

#pragma unroll
  for (int j = 0; j < 4; ++j) {
    const int col = colBase + j * 16 + ml;
    const int r0 = rowTile + hlf * 8;
#pragma unroll
    for (int v = 0; v < 8; ++v)
      sim[(size_t)(r0 + v) * MROWS + col] = acc[j][v];
  }
}

// ---------------------------------------------------------------------------
// Kernel 8: NT-Xent reduction; thread r owns row r of sim[256,256].
// loss_r = log(sum_{c!=r} exp(2*sim)) - 2*sim[r][(r+128)%256]; out = mean.
// ---------------------------------------------------------------------------
__global__ __launch_bounds__(256) void ntxent_loss(const float* __restrict__ sim,
                                                   float* __restrict__ out) {
  __shared__ float red[256];
  const int r = threadIdx.x;
  float denom = 0.0f;
  for (int c = 0; c < MROWS; ++c) {
    float e = __expf(sim[r * MROWS + c] * T_INV);
    denom += (c == r) ? 0.0f : e;
  }
  const float pos = sim[r * MROWS + ((r + 128) & 255)];
  red[r] = __logf(denom) - pos * T_INV;
  __syncthreads();
  for (int st = 128; st > 0; st >>= 1) {
    if (r < st) red[r] += red[r + st];
    __syncthreads();
  }
  if (r == 0) out[0] = red[0] / (float)MROWS;
}

// ---------------------------------------------------------------------------
extern "C" void kernel_launch(void* const* d_in, const int* in_sizes, int n_in,
                              void* d_out, int out_size, void* d_ws, size_t ws_size,
                              hipStream_t stream) {
  (void)in_sizes; (void)n_in; (void)out_size;
  const float* ct  = (const float*)d_in[0];
  const float* mct = (const float*)d_in[1];
  const float* W1  = (const float*)d_in[2];
  const float* b1  = (const float*)d_in[3];
  const float* W2  = (const float*)d_in[4];
  const float* b2  = (const float*)d_in[5];
  float* out = (float*)d_out;
  char* ws = (char*)d_ws;

  // ---- workspace layout (byte offsets, all 16B aligned) -------------------
  const size_t szXpl   = (size_t)MROWS * DIN * sizeof(__bf16);        // 7,864,320
  const size_t szHpart = (size_t)SPLITK * MROWS * N1 * sizeof(float); // 16,777,216
  const size_t szHpl   = (size_t)MROWS * N1 * sizeof(__bf16);         // 2,097,152
  const size_t szR     = (size_t)MROWS * N2 * sizeof(float);          // 2,097,152

  size_t off = 0;
  __bf16* Xhi = (__bf16*)(ws + off); off += szXpl;
  __bf16* Xlo = (__bf16*)(ws + off); off += szXpl;
  const size_t partOff = off;                      // Hpart, later reused by Rpart
  float*  Hpart = (float*)(ws + partOff); off += szHpart;
  __bf16* Hhi = (__bf16*)(ws + off); off += szHpl;
  __bf16* Hlo = (__bf16*)(ws + off); off += szHpl;
  float*  R   = (float*)(ws + off); off += szR;
  float*  inv = (float*)(ws + off); off += 1024;
  float*  sim = (float*)(ws + off); off += (size_t)MROWS * MROWS * sizeof(float);
  float*  Rpart = (float*)(ws + partOff);          // overlaps dead Hpart

  if (ws_size < off) return;  // deterministic no-op if workspace too small

  build_X_planes<<<dim3((MROWS * DIN) / 256), dim3(256), 0, stream>>>(ct, mct, Xhi, Xlo);

  wmma_gemm_splitk<<<dim3(N1 / 128, SPLITK), dim3(256), 0, stream>>>(
      Xhi, Xlo, W1, Hpart, DIN, N1, DIN / SPLITK);
  reduce_bias_relu<<<dim3((MROWS * N1) / 256), dim3(256), 0, stream>>>(
      Hpart, b1, Hhi, Hlo, nullptr, N1);

  wmma_gemm_splitk<<<dim3(N2 / 128, SPLITK), dim3(256), 0, stream>>>(
      Hhi, Hlo, W2, Rpart, N1, N2, N1 / SPLITK);
  reduce_bias_relu<<<dim3((MROWS * N2) / 256), dim3(256), 0, stream>>>(
      Rpart, b2, nullptr, nullptr, R, N2);

  row_inv_norms<<<dim3(MROWS), dim3(256), 0, stream>>>(R, inv);
  wmma_gram<<<dim3(MROWS / 64, 2), dim3(256), 0, stream>>>(R, inv, sim);
  ntxent_loss<<<dim3(1), dim3(256), 0, stream>>>(sim, out);
}
